// VectorQuantizer_29463475651259
// MI455X (gfx1250) — matrix operand order repaired and checked by reading, not compiled, not run on hardware
//
#include <hip/hip_runtime.h>
#include <hip/hip_bf16.h>
#include <stdint.h>
#include <stddef.h>

// Problem constants (match reference)
#define B_   16
#define D_   512
#define HW_  1024
#define N_   (B_ * HW_)   // 16384 tokens
#define CD_  256          // codebook dim
#define K_   8192         // num codes

typedef __attribute__((ext_vector_type(16))) __bf16 v16bf;
typedef __attribute__((ext_vector_type(8)))  float  v8f;

union FragBf { uint4 u[2]; v16bf v; };

// ---------------------------------------------------------------------------
// Fragment loaders matching CDNA5 WMMA bf16 16x16x32 VGPR layouts (ISA 7.12.2)
// A (16x32, MxK): lanes 0-15 hold M=lane, K=[k..k+7] then [k+16..k+23];
//                 lanes 16-31 hold M=lane-16, K=[k+8..k+15] then [k+24..k+31].
// B (32x16, KxN): lane holds column N=lane%16; lanes 0-15 K=[k..k+7] then [k+16..k+23],
//                 lanes 16-31 K=[k+8..k+15] then [k+24..k+31] (same as A).
// ---------------------------------------------------------------------------
__device__ __forceinline__ v16bf load_frag_a(const __bf16* rowptr, int kbase, int lane) {
  const int sel = (lane < 16) ? 0 : 8;
  FragBf f;
  f.u[0] = *(const uint4*)(rowptr + kbase + sel);
  f.u[1] = *(const uint4*)(rowptr + kbase + sel + 16);
  return f.v;
}
__device__ __forceinline__ v16bf load_frag_b(const __bf16* colptr, int kbase, int lane) {
  const int off = (lane < 16) ? 0 : 8;
  FragBf f;
  f.u[0] = *(const uint4*)(colptr + kbase + off);
  f.u[1] = *(const uint4*)(colptr + kbase + off + 16);
  return f.v;
}

// ---------------------------------------------------------------------------
// Prep kernels (bandwidth trivial vs 23.3 TB/s)
// ---------------------------------------------------------------------------
__global__ void k_cvt_bf16(const float* __restrict__ src, __bf16* __restrict__ dst, int n) {
  int i = blockIdx.x * blockDim.x + threadIdx.x;
  if (i < n) dst[i] = (__bf16)src[i];
}

__global__ void k_cbnorm(const float* __restrict__ cb, float* __restrict__ cbn) {
  __shared__ float sred[8];
  const int k = blockIdx.x, c = threadIdx.x;       // 256 threads, one per CD elem
  float v = cb[(size_t)k * CD_ + c];
  float s = v * v;
  for (int m = 16; m >= 1; m >>= 1) s += __shfl_xor(s, m, 32);
  if ((c & 31) == 0) sred[c >> 5] = s;
  __syncthreads();
  if (c == 0) {
    float t = 0.f;
    for (int i = 0; i < 8; ++i) t += sred[i];
    cbn[k] = t;
  }
}

// x [B, D, HW] f32  ->  xt [N=B*HW, D] bf16   (32x32 LDS tile transpose)
__global__ void k_transpose_x(const float* __restrict__ x, __bf16* __restrict__ xt) {
  __shared__ float tile[32][33];
  const int b = blockIdx.z, dt = blockIdx.y, ht = blockIdx.x;
  const int j = threadIdx.x & 31;
  const int i = threadIdx.x >> 5;                   // 0..7
  const float* src = x + (size_t)b * D_ * HW_ + (size_t)(dt * 32) * HW_ + ht * 32;
  for (int ii = 0; ii < 32; ii += 8)
    tile[i + ii][j] = src[(size_t)(i + ii) * HW_ + j];   // coalesced along hw
  __syncthreads();
  __bf16* dst = xt + (size_t)(b * HW_ + ht * 32) * D_ + dt * 32;
  for (int ii = 0; ii < 32; ii += 8)
    dst[(size_t)(i + ii) * D_ + j] = (__bf16)tile[j][i + ii]; // coalesced along d
}

__global__ void k_zero(float* p) { p[0] = 0.f; }

// ---------------------------------------------------------------------------
// GEMM 1: z[N, CD] = xt[N, D] @ W_in^T + b_in   (bf16 WMMA, f32 accum)
// grid (1024, 4), block 128 (4 waves); one wave per 16x16 tile, K loop = 16.
// ---------------------------------------------------------------------------
__global__ void k_gemm_in(const __bf16* __restrict__ xt, const __bf16* __restrict__ Win,
                          const float* __restrict__ b_in,
                          float* __restrict__ z, __bf16* __restrict__ zb) {
  const int lane = threadIdx.x & 31, wave = threadIdx.x >> 5;
  const int r0 = blockIdx.x * 16;
  const int c0 = (blockIdx.y * 4 + wave) * 16;
  const int nl = lane & 15, hi = lane >> 4;
  const float bias = b_in[c0 + nl];
  v8f acc;
  for (int g = 0; g < 8; ++g) acc[g] = bias;        // bias: same N per lane, all M
  const __bf16* arow = xt  + (size_t)(r0 + nl) * D_;
  const __bf16* bcol = Win + (size_t)(c0 + nl) * D_; // B(d,c)=W_in[c,d]: contiguous d
  for (int k0 = 0; k0 < D_; k0 += 32) {
    v16bf a = load_frag_a(arow, k0, lane);
    v16bf b = load_frag_b(bcol, k0, lane);
    acc = __builtin_amdgcn_wmma_f32_16x16x32_bf16(false, a, false, b, (short)0, acc, false, false);
  }
  for (int g = 0; g < 8; ++g) {
    const size_t o = (size_t)(r0 + g + 8 * hi) * CD_ + (c0 + nl);
    z[o]  = acc[g];
    zb[o] = (__bf16)acc[g];
  }
}

// ---------------------------------------------------------------------------
// Dominant kernel: argmin_k ( ||e_k||^2 - 2 z.e_k )  via bf16 WMMA with a
// double-buffered GLOBAL_LOAD_ASYNC_TO_LDS_B128 pipeline (ASYNCcnt).
//
// grid (128, 2), block 256 (8 waves). Each wave owns 16 rows (A preloaded into
// 8 register fragments); all 8 waves share each staged 32-code LDS chunk
// (halves L2 codebook traffic vs 4-wave blocks). gridDim.y splits the code
// range in two; per-row (minval,minidx) candidates merged by k_merge.
// LDS pitch 264 bf16 -> lane bank stride 4 -> conflict-free B fragment reads.
// Dynamic LDS only (offset 0), 2 x 32 x 264 bf16 = 33792 B.
// ---------------------------------------------------------------------------
#define CCHUNK 32
#define PITCH  264
#define BUFB   (CCHUNK * PITCH * 2)   // bytes per LDS buffer

extern __shared__ __bf16 smem_dyn[];

__device__ __forceinline__ void stage_chunk_async(const __bf16* __restrict__ cb,
                                                  int codebase, uint32_t ldsbase,
                                                  int tid) {
  // 32 codes x 256 bf16 = 1024 x 16B units; 256 threads -> 4 units each.
  // Per-lane async copy: 16 B global -> LDS, tracked by ASYNCcnt.
  for (int u = tid; u < CCHUNK * 32; u += 256) {
    const int row = u >> 5, off16 = (u & 31);
    const uint32_t lds = ldsbase + (uint32_t)(row * (PITCH * 2) + off16 * 16);
    const __bf16* g = cb + (size_t)(codebase + row) * CD_ + off16 * 8;
    asm volatile("global_load_async_to_lds_b128 %0, %1, off"
                 :: "v"(lds), "v"(g) : "memory");
  }
}

__global__ void k_argmin(const __bf16* __restrict__ zb, const __bf16* __restrict__ cb,
                         const float* __restrict__ cbn,
                         float* __restrict__ mv_out, int* __restrict__ mi_out) {
  const int tid = threadIdx.x;                      // 0..255
  const int lane = tid & 31, wave = tid >> 5;       // 8 waves
  const int r0 = blockIdx.x * 128 + wave * 16;
  const int nl = lane & 15, hi = lane >> 4;
  const int half = blockIdx.y;                      // code-range split
  const int cstart = half * (K_ / 2);
  const int nchunks = (K_ / 2) / CCHUNK;            // 128

  v16bf afr[8];
  const __bf16* arow = zb + (size_t)(r0 + nl) * CD_;
  for (int f = 0; f < 8; ++f) afr[f] = load_frag_a(arow, f * 32, lane);

  float mv[8]; int mi[8];
  for (int g = 0; g < 8; ++g) { mv[g] = 3.4e38f; mi[g] = 0; }

  // prologue: stage chunk 0
  stage_chunk_async(cb, cstart, 0u, tid);
  asm volatile("s_wait_asynccnt 0x0" ::: "memory");
  __syncthreads();

  for (int ch = 0; ch < nchunks; ++ch) {
    const int cc0 = cstart + ch * CCHUNK;
    // kick off next chunk into the other buffer (overlaps with WMMA below);
    // safe: the barrier at the end of the previous iteration guarantees all
    // waves finished computing on that buffer.
    if (ch + 1 < nchunks)
      stage_chunk_async(cb, cc0 + CCHUNK, (uint32_t)(((ch + 1) & 1) * BUFB), tid);

    const __bf16* sbuf = smem_dyn + (ch & 1) * (CCHUNK * PITCH);
    for (int s = 0; s < CCHUNK / 16; ++s) {
      v8f acc = {};
      const __bf16* bcol = sbuf + (s * 16 + nl) * PITCH;
      for (int kk = 0; kk < 8; ++kk) {
        v16bf b = load_frag_b(bcol, kk * 32, lane);
        acc = __builtin_amdgcn_wmma_f32_16x16x32_bf16(false, afr[kk], false, b, (short)0, acc, false, false);
      }
      const float nrm = cbn[cc0 + s * 16 + nl];     // L2-hot, 16 consecutive dwords
      const int code = cc0 + s * 16 + nl;
      for (int g = 0; g < 8; ++g) {
        const float d2 = nrm - 2.0f * acc[g];
        if (d2 < mv[g]) { mv[g] = d2; mi[g] = code; } // strict '<': first index wins
      }
    }
    asm volatile("s_wait_asynccnt 0x0" ::: "memory"); // next buffer landed
    __syncthreads();                                   // visible to all waves
  }

  // cross-lane argmin over the 16 columns held by each 16-lane half
  for (int g = 0; g < 8; ++g) {
    float v = mv[g]; int i = mi[g];
    for (int m = 1; m <= 8; m <<= 1) {
      const float ov = __shfl_xor(v, m, 32);
      const int   oi = __shfl_xor(i, m, 32);
      if (ov < v || (ov == v && oi < i)) { v = ov; i = oi; }
    }
    if (nl == 0) {
      const int row = r0 + g + 8 * hi;
      mv_out[half * N_ + row] = v;
      mi_out[half * N_ + row] = i;
    }
  }
}

// Merge the two code-range halves per row (first-index tie-break).
__global__ void k_merge(const float* __restrict__ mv2, const int* __restrict__ mi2,
                        int* __restrict__ idx, float* __restrict__ idx_f) {
  const int i = blockIdx.x * blockDim.x + threadIdx.x;
  if (i >= N_) return;
  const float v0 = mv2[i], v1 = mv2[N_ + i];
  const int   i0 = mi2[i], i1 = mi2[N_ + i];
  const int best = (v1 < v0 || (v1 == v0 && i1 < i0)) ? i1 : i0;
  idx[i]   = best;
  idx_f[i] = (float)best;
}

// ---------------------------------------------------------------------------
// GEMM 2: out = codebook[idx] @ W_out^T + b_out, stored directly as [B,D,H,W].
// Each lane's 8 C values are 8 consecutive hw at fixed d -> two float4 stores.
// grid (1024, 8), block 128.
// ---------------------------------------------------------------------------
__global__ void k_gemm_out(const int* __restrict__ idx, const __bf16* __restrict__ cb,
                           const __bf16* __restrict__ Wout, const float* __restrict__ b_out,
                           float* __restrict__ out) {
  const int lane = threadIdx.x & 31, wave = threadIdx.x >> 5;
  const int r0 = blockIdx.x * 16;
  const int d0 = (blockIdx.y * 4 + wave) * 16;
  const int nl = lane & 15, hi = lane >> 4;
  const int code = idx[r0 + nl];
  const __bf16* arow = cb   + (size_t)code * CD_;          // gathered A row
  const __bf16* bcol = Wout + (size_t)(d0 + nl) * CD_;     // B(c,d)=W_out[d,c]
  const float bias = b_out[d0 + nl];
  v8f acc;
  for (int g = 0; g < 8; ++g) acc[g] = bias;
  for (int k0 = 0; k0 < CD_; k0 += 32) {
    v16bf a = load_frag_a(arow, k0, lane);
    v16bf b = load_frag_b(bcol, k0, lane);
    acc = __builtin_amdgcn_wmma_f32_16x16x32_bf16(false, a, false, b, (short)0, acc, false, false);
  }
  const int bimg = r0 / HW_;
  const int hw0  = (r0 % HW_) + 8 * hi;                    // 16-row tiles never cross a batch
  float* p = out + (size_t)bimg * D_ * HW_ + (size_t)(d0 + nl) * HW_ + hw0;
  *(float4*)(p)     = make_float4(acc[0], acc[1], acc[2], acc[3]);
  *(float4*)(p + 4) = make_float4(acc[4], acc[5], acc[6], acc[7]);
}

// ---------------------------------------------------------------------------
// Commitment loss: mean((q - z)^2) in f32, q gathered from original codebook.
// ---------------------------------------------------------------------------
__global__ void k_loss(const int* __restrict__ idx, const float* __restrict__ cb,
                       const float* __restrict__ z, float* __restrict__ accum) {
  __shared__ float sred[8];
  const int n = blockIdx.x, c = threadIdx.x;        // 256 threads per row
  const float d = cb[(size_t)idx[n] * CD_ + c] - z[(size_t)n * CD_ + c];
  float s = d * d;
  for (int m = 16; m >= 1; m >>= 1) s += __shfl_xor(s, m, 32);
  if ((c & 31) == 0) sred[c >> 5] = s;
  __syncthreads();
  if (c == 0) {
    float t = 0.f;
    for (int i = 0; i < 8; ++i) t += sred[i];
    atomicAdd(accum, t);
  }
}

__global__ void k_final(const float* __restrict__ accum, float* __restrict__ loss_out) {
  loss_out[0] = accum[0] * (1.0f / ((float)N_ * (float)CD_));  // COMMIT = 1.0
}

// ---------------------------------------------------------------------------
// Launch
// ---------------------------------------------------------------------------
extern "C" void kernel_launch(void* const* d_in, const int* in_sizes, int n_in,
                              void* d_out, int out_size, void* d_ws, size_t ws_size,
                              hipStream_t stream) {
  (void)in_sizes; (void)n_in; (void)out_size;
  const float* x     = (const float*)d_in[0];
  const float* W_in  = (const float*)d_in[1];
  const float* b_in  = (const float*)d_in[2];
  const float* W_out = (const float*)d_in[3];
  const float* b_out = (const float*)d_in[4];
  const float* cb    = (const float*)d_in[5];

  // Workspace layout (256-B aligned), ~47 MB total
  char* ws = (char*)d_ws;
  size_t o = 0;
  auto alloc = [&](size_t bytes) { char* p = ws + o; o = (o + bytes + 255) & ~(size_t)255; return p; };
  __bf16* xt_bf   = (__bf16*)alloc((size_t)N_ * D_ * 2);     // 16 MB
  __bf16* Win_bf  = (__bf16*)alloc((size_t)CD_ * D_ * 2);
  __bf16* Wout_bf = (__bf16*)alloc((size_t)D_ * CD_ * 2);
  __bf16* cb_bf   = (__bf16*)alloc((size_t)K_ * CD_ * 2);    // 4 MB
  float*  cb_n    = (float*) alloc((size_t)K_ * 4);
  float*  z_f32   = (float*) alloc((size_t)N_ * CD_ * 4);    // 16 MB
  __bf16* z_bf    = (__bf16*)alloc((size_t)N_ * CD_ * 2);    // 8 MB
  int*    idx_i   = (int*)   alloc((size_t)N_ * 4);
  float*  mv2     = (float*) alloc((size_t)2 * N_ * 4);      // per-half min value
  int*    mi2     = (int*)   alloc((size_t)2 * N_ * 4);      // per-half min index
  float*  acc     = (float*) alloc(256);
  (void)ws_size;

  float* out_f   = (float*)d_out;                            // [B,D,H,W]
  float* out_idx = out_f + (size_t)B_ * D_ * HW_;            // indices as float
  float* out_ls  = out_idx + N_;                             // scalar loss

  // 1) prep
  k_cvt_bf16<<<(CD_ * D_ + 255) / 256, 256, 0, stream>>>(W_in, Win_bf, CD_ * D_);
  k_cvt_bf16<<<(D_ * CD_ + 255) / 256, 256, 0, stream>>>(W_out, Wout_bf, D_ * CD_);
  k_cvt_bf16<<<(K_ * CD_ + 255) / 256, 256, 0, stream>>>(cb, cb_bf, K_ * CD_);
  k_cbnorm<<<K_, 256, 0, stream>>>(cb, cb_n);
  k_transpose_x<<<dim3(HW_ / 32, D_ / 32, B_), 256, 0, stream>>>(x, xt_bf);
  k_zero<<<1, 1, 0, stream>>>(acc);

  // 2) project in (WMMA bf16)
  k_gemm_in<<<dim3(N_ / 16, 4), 128, 0, stream>>>(xt_bf, Win_bf, b_in, z_f32, z_bf);

  // 3) nearest-code argmin (dominant WMMA kernel, async-LDS double buffering)
  k_argmin<<<dim3(N_ / 128, 2), 256, 2 * BUFB, stream>>>(z_bf, cb_bf, cb_n, mv2, mi2);
  k_merge<<<(N_ + 255) / 256, 256, 0, stream>>>(mv2, mi2, idx_i, out_idx);

  // 4) project out with gathered codes (WMMA bf16), direct [B,D,H,W] stores
  k_gemm_out<<<dim3(N_ / 16, 8), 128, 0, stream>>>(idx_i, cb_bf, Wout_bf, b_out, out_f);

  // 5) commitment loss
  k_loss<<<N_, 256, 0, stream>>>(idx_i, cb, z_f32, acc);
  k_final<<<1, 1, 0, stream>>>(acc, out_ls);
}
